// ConditionalMamba_40407052320986
// MI455X (gfx1250) — matrix operand order repaired and checked by reading, not compiled, not run on hardware
//
#include <hip/hip_runtime.h>
#include <hip/hip_bf16.h>
#include <math.h>

typedef __attribute__((ext_vector_type(16))) _Float16 v16h;
typedef __attribute__((ext_vector_type(8)))  _Float16 v8h;
typedef __attribute__((ext_vector_type(8)))  float    v8f;

#define BATCH  2
#define CCH    64
#define HH     64
#define WW     64
#define HWN    4096
#define LTOT   8192
#define DINNER 128
#define DSTATE 16
#define DTRANK 4
#define NCHUNK 64
#define CHUNKT 128   /* 8192 / 64 */

static __device__ __forceinline__ v8f wmma16x16x32(v16h a, v16h b, v8f c) {
  // v_wmma_f32_16x16x32_f16: D = A(16x32 f16) * B(32x16 f16) + C(f32)
  return __builtin_amdgcn_wmma_f32_16x16x32_f16(false, a, false, b, (short)0, c,
                                                false, false);
}

// Build a v16h fragment from two contiguous, 16B-aligned 8-half runs (LDS or
// global).  Lowered to wide loads (ds_load_b128 / global b128) by the compiler.
static __device__ __forceinline__ v16h ld2x8(const _Float16* p0,
                                             const _Float16* p1) {
  const v8h a = *(const v8h*)p0;
  const v8h c = *(const v8h*)p1;
  v16h r;
  #pragma unroll
  for (int i = 0; i < 8; ++i) { r[i] = a[i]; r[i + 8] = c[i]; }
  return r;
}
static __device__ __forceinline__ v16h ld16(const _Float16* p) {
  return ld2x8(p, p + 8);
}

// ---------------------------------------------------------------------------
// 3x3 SAME conv + leaky_relu(0.01) as LDS-staged implicit GEMM.
// Block (256 thr / 8 waves) = one (batch, row y, 32 c_out) strip.
// LDS: input slab f16 [3][66][64] (zero halos), weights f16 [32][9][64].
// Each wave: 16 c_out x 16 x tile, K = 9 taps x 64 cin -> 18 WMMA.
// ---------------------------------------------------------------------------
__global__ void conv3x3_leaky_wmma(const float* __restrict__ in,
                                   const float* __restrict__ w,
                                   const float* __restrict__ bias,
                                   float* __restrict__ out) {
  __shared__ _Float16 s_in[3 * 66 * 64];   // [row][x+1][cin]
  __shared__ _Float16 s_w[32 * 9 * 64];    // [co_local][tap][cin]

  const int tid  = threadIdx.x;
  const int y    = blockIdx.x & 63;
  const int cotg = (blockIdx.x >> 6) & 1;  // which 32-channel half of c_out
  const int b    = blockIdx.x >> 7;

  // Stage input rows y-1..y+1 (zero padded rows), coalesced along x.
  for (int idx = tid; idx < 3 * 64 * 64; idx += 256) {
    const int x = idx & 63;
    const int c = (idx >> 6) & 63;
    const int r = idx >> 12;
    const int yy = y + r - 1;
    float v = (yy >= 0 && yy < HH)
                  ? in[((size_t)b * CCH + c) * HWN + yy * WW + x] : 0.f;
    s_in[(r * 66 + x + 1) * 64 + c] = (_Float16)v;
  }
  // Zero halo columns x = -1 and x = 64.
  for (int idx = tid; idx < 2 * 3 * 64; idx += 256) {
    const int side = (idx >= 3 * 64) ? 1 : 0;
    const int rem  = idx - side * 3 * 64;
    const int r = rem >> 6;
    const int c = rem & 63;
    s_in[(r * 66 + (side ? 65 : 0)) * 64 + c] = (_Float16)0.f;
  }
  // Stage 32 c_out of weights; iterate in global (OIHW) order -> coalesced.
  const float* wbase = w + (size_t)(cotg * 32) * (CCH * 9);
  for (int g = tid; g < 32 * CCH * 9; g += 256) {
    const int co  = g / 576;
    const int rem = g - co * 576;
    const int cin = rem / 9;
    const int tap = rem - cin * 9;
    s_w[(co * 9 + tap) * 64 + cin] = (_Float16)wbase[g];
  }
  __syncthreads();

  const int lane = tid & 31;
  const int wv   = tid >> 5;        // 8 waves
  const int xt   = wv & 3;          // x tile
  const int col  = (wv >> 2) * 16;  // c_out tile within the 32
  const int mn   = lane & 15;       // A row (m) == B col (n)
  const int hi   = lane >> 4;

  v8f acc = {};
  for (int tap = 0; tap < 9; ++tap) {
    const int dy = tap / 3, dx = tap % 3;            // 0..2
    const int xsh = xt * 16 + mn + dx;               // padded x index 0..65
    const _Float16* prow = &s_in[(dy * 66 + xsh) * 64];
    const _Float16* pw   = &s_w[((col + mn) * 9 + tap) * 64];
    #pragma unroll
    for (int kc = 0; kc < CCH; kc += 32) {
      // A 16x32 f16: K = h + 8*(h>=8) + 8*(lane>=16) -> two 8-half runs
      v16h af = ld2x8(pw + kc + hi * 8, pw + kc + 16 + hi * 8);
      // B 32x16 f16: K = h + 8*(h>=8) + 8*(lane>=16) -> two 8-half runs
      v16h bf = ld2x8(prow + kc + hi * 8, prow + kc + 16 + hi * 8);
      acc = wmma16x16x32(af, bf, acc);
    }
  }
  #pragma unroll
  for (int i = 0; i < 8; ++i) {
    const int co = cotg * 32 + col + i + (hi ? 8 : 0);
    float v = acc[i] + bias[co];
    v = (v > 0.f) ? v : 0.01f * v;
    out[((size_t)b * CCH + co) * HWN + y * WW + xt * 16 + mn] = v;
  }
}

// ---------------------------------------------------------------------------
// in_proj: xz[b,l,0:256] = x[b,l,0:64] @ in_proj_w.T
// Block stages one 16-token x 64-ch B tile in LDS (f16, [t][k]); its 8 waves
// each produce a different 16-row m tile (2 blocks cover all 256 rows).
// ---------------------------------------------------------------------------
__global__ void in_proj_wmma_k(const float* __restrict__ cond,
                               const float* __restrict__ prim,
                               const float* __restrict__ ipw,
                               float* __restrict__ xz) {
  __shared__ _Float16 s_b[16 * 64];        // [t_local][k]
  const int tid = threadIdx.x;
  const int nt  = blockIdx.x & 511;        // 512 token tiles
  const int mtg = (blockIdx.x >> 9) & 1;   // row-tile group (8 tiles each)
  const int b   = blockIdx.x >> 10;
  const int tbase = nt * 16;               // tile never straddles 4096
  const float* src = (tbase < HWN) ? (cond + (size_t)b * CCH * HWN)
                                   : (prim + (size_t)b * CCH * HWN);
  const int tb = (tbase < HWN) ? tbase : (tbase - HWN);

  for (int idx = tid; idx < 16 * 64; idx += 256) {
    const int tl = idx & 15;
    const int k  = idx >> 4;
    s_b[tl * 64 + k] = (_Float16)src[(size_t)k * HWN + tb + tl];
  }
  __syncthreads();

  const int lane = tid & 31;
  const int mt   = mtg * 8 + (tid >> 5);   // 16 row tiles total (256 rows)
  const int ln   = lane & 15;
  const int hi   = lane >> 4;
  const float* arow = ipw + (size_t)(mt * 16 + ln) * 64;

  v8f acc = {};
  #pragma unroll
  for (int kc = 0; kc < 64; kc += 32) {
    v16h af;
    #pragma unroll
    for (int h = 0; h < 8; ++h) {          // contiguous f32 runs -> wide loads
      af[h]     = (_Float16)arow[kc + hi * 8 + h];
      af[h + 8] = (_Float16)arow[kc + 16 + hi * 8 + h];
    }
    v16h bf = ld2x8(&s_b[ln * 64 + kc + hi * 8], &s_b[ln * 64 + kc + 16 + hi * 8]);
    acc = wmma16x16x32(af, bf, acc);
  }
  #pragma unroll
  for (int i = 0; i < 8; ++i) {
    const int m = mt * 16 + i + (hi ? 8 : 0);
    xz[((size_t)b * LTOT + tbase + ln) * 256 + m] = acc[i];
  }
}

// ---------------------------------------------------------------------------
// Depthwise causal conv1d (K=4) + SiLU: xc[b,l,d]
// ---------------------------------------------------------------------------
__global__ void dwconv_silu(const float* __restrict__ xz,
                            const float* __restrict__ w1d,
                            const float* __restrict__ b1d,
                            float* __restrict__ xc) {
  int tid = blockIdx.x * blockDim.x + threadIdx.x;
  if (tid >= BATCH * LTOT * DINNER) return;
  int d = tid & 127;
  int l = (tid >> 7) & (LTOT - 1);
  int b = tid >> 20;
  float acc = b1d[d];
  #pragma unroll
  for (int j = 0; j < 4; ++j) {
    int ll = l - 3 + j;
    if (ll >= 0) acc += w1d[d * 4 + j] * xz[((size_t)b * LTOT + ll) * 256 + d];
  }
  xc[((size_t)b * LTOT + l) * DINNER + d] = acc / (1.f + __expf(-acc));
}

// ---------------------------------------------------------------------------
// x_proj (128 -> 36) + dt_proj (4 -> 128) + softplus, one token per block.
// ---------------------------------------------------------------------------
__global__ void xproj_dt(const float* __restrict__ xc,
                         const float* __restrict__ xpw,
                         const float* __restrict__ dtpw,
                         const float* __restrict__ dtpb,
                         float* __restrict__ dt_out,
                         float* __restrict__ Bm,
                         float* __restrict__ Cm) {
  __shared__ float s_xc[DINNER];
  __shared__ float s_dbl[DTRANK];
  const int tok = blockIdx.x;   // b*LTOT + l
  const int tid = threadIdx.x;  // 128
  s_xc[tid] = xc[(size_t)tok * DINNER + tid];
  __syncthreads();
  if (tid < DTRANK + 2 * DSTATE) {
    float acc = 0.f;
    const float* row = xpw + tid * DINNER;
    for (int k = 0; k < DINNER; ++k) acc += row[k] * s_xc[k];
    if (tid < DTRANK)               s_dbl[tid] = acc;
    else if (tid < DTRANK + DSTATE) Bm[(size_t)tok * DSTATE + (tid - DTRANK)] = acc;
    else                            Cm[(size_t)tok * DSTATE + (tid - DTRANK - DSTATE)] = acc;
  }
  __syncthreads();
  float v = dtpb[tid];
  #pragma unroll
  for (int r = 0; r < DTRANK; ++r) v += dtpw[tid * DTRANK + r] * s_dbl[r];
  v = (v > 20.f) ? v : log1pf(__expf(v));  // softplus
  dt_out[(size_t)tok * DINNER + tid] = v;
}

// ---------------------------------------------------------------------------
// Scan pass 1: per-chunk affine summary (h_out = aP*h_in + u) per (b,d,n).
// Block = 256 threads = 16 d x 16 n.
// ---------------------------------------------------------------------------
__global__ void scan_pass1(const float* __restrict__ dt,
                           const float* __restrict__ xc,
                           const float* __restrict__ Bm,
                           const float* __restrict__ A_log,
                           float* __restrict__ aP_out,
                           float* __restrict__ u_out) {
  const int dg = blockIdx.x & 7;
  const int s  = (blockIdx.x >> 3) & 63;
  const int b  = blockIdx.x >> 9;
  const int n  = threadIdx.x & 15;
  const int d  = dg * 16 + (threadIdx.x >> 4);
  const float A = -__expf(A_log[d * DSTATE + n]);
  float aP = 1.f, u = 0.f;
  const int l0 = s * CHUNKT;
  for (int i = 0; i < CHUNKT; ++i) {
    const size_t base = (size_t)b * LTOT + (l0 + i);
    const float dtv = dt[base * DINNER + d];
    const float xcv = xc[base * DINNER + d];
    const float Bv  = Bm[base * DSTATE + n];
    const float a = __expf(dtv * A);
    u = a * u + dtv * Bv * xcv;
    aP *= a;
  }
  const size_t idx = (((size_t)b * NCHUNK + s) * DINNER + d) * DSTATE + n;
  aP_out[idx] = aP;
  u_out[idx] = u;
}

// ---------------------------------------------------------------------------
// Scan pass 2: serial combine of the 64 chunk summaries -> h at chunk starts.
// ---------------------------------------------------------------------------
__global__ void scan_pass2(const float* __restrict__ aP,
                           const float* __restrict__ u,
                           float* __restrict__ hstart) {
  int tid = blockIdx.x * blockDim.x + threadIdx.x;
  if (tid >= BATCH * DINNER * DSTATE) return;
  int n = tid & 15;
  int d = (tid >> 4) & 127;
  int b = tid >> 11;
  float h = 0.f;
  for (int s = 0; s < NCHUNK; ++s) {
    size_t idx = (((size_t)b * NCHUNK + s) * DINNER + d) * DSTATE + n;
    hstart[idx] = h;
    h = aP[idx] * h + u[idx];
  }
}

// ---------------------------------------------------------------------------
// Scan pass 3: replay chunks s >= 32 (only l >= 4096 produces output),
// reduce over n with shfl_xor (width 16), fuse (y + xc*D) * silu(z).
// ---------------------------------------------------------------------------
__global__ void scan_pass3(const float* __restrict__ dt,
                           const float* __restrict__ xc,
                           const float* __restrict__ xz,
                           const float* __restrict__ Bm,
                           const float* __restrict__ Cm,
                           const float* __restrict__ A_log,
                           const float* __restrict__ Dp,
                           const float* __restrict__ hstart,
                           float* __restrict__ yf) {
  const int dg = blockIdx.x & 7;
  const int s  = ((blockIdx.x >> 3) & 31) + 32;
  const int b  = blockIdx.x >> 8;
  const int n  = threadIdx.x & 15;
  const int d  = dg * 16 + (threadIdx.x >> 4);
  const float A  = -__expf(A_log[d * DSTATE + n]);
  const float Dv = Dp[d];
  float h = hstart[(((size_t)b * NCHUNK + s) * DINNER + d) * DSTATE + n];
  const int l0 = s * CHUNKT;
  for (int i = 0; i < CHUNKT; ++i) {
    const int l = l0 + i;
    const size_t base = (size_t)b * LTOT + l;
    const float dtv = dt[base * DINNER + d];
    const float xcv = xc[base * DINNER + d];
    const float Bv  = Bm[base * DSTATE + n];
    const float Cv  = Cm[base * DSTATE + n];
    const float a = __expf(dtv * A);
    h = a * h + dtv * Bv * xcv;
    float r = h * Cv;
    r += __shfl_xor(r, 1, 16);
    r += __shfl_xor(r, 2, 16);
    r += __shfl_xor(r, 4, 16);
    r += __shfl_xor(r, 8, 16);
    if (n == 0) {
      float z = xz[base * 256 + DINNER + d];
      float yv = (r + xcv * Dv) * (z / (1.f + __expf(-z)));
      yf[((size_t)b * HWN + (l - HWN)) * DINNER + d] = yv;
    }
  }
}

// ---------------------------------------------------------------------------
// out_proj (128 -> 64) over the last 4096 tokens; writes NCHW output.
// yf is [token][k] (k contiguous) and opw rows are contiguous, so both
// fragment gathers are wide contiguous loads already.
// ---------------------------------------------------------------------------
__global__ void out_proj_wmma_k(const float* __restrict__ yf,
                                const float* __restrict__ opw,
                                float* __restrict__ out) {
  const int wave = blockIdx.x * (blockDim.x >> 5) + (threadIdx.x >> 5);
  const int lane = threadIdx.x & 31;
  const int nt = wave & 255;        // 256 token tiles of 16
  const int mt = (wave >> 8) & 3;   // 4 row tiles -> 64 channels
  const int b  = wave >> 10;
  const int ln = lane & 15;
  const int hi = lane >> 4;
  const int t  = nt * 16 + ln;      // hw index
  const float* arow = opw + (size_t)(mt * 16 + ln) * DINNER;
  const float* brow = yf + ((size_t)b * HWN + t) * DINNER;
  v8f acc = {};
  #pragma unroll
  for (int kc = 0; kc < DINNER; kc += 32) {
    v16h af, bf;
    #pragma unroll
    for (int h = 0; h < 8; ++h) {
      af[h]     = (_Float16)arow[kc + hi * 8 + h];
      af[h + 8] = (_Float16)arow[kc + 16 + hi * 8 + h];
      bf[h]     = (_Float16)brow[kc + hi * 8 + h];
      bf[h + 8] = (_Float16)brow[kc + 16 + hi * 8 + h];
    }
    acc = wmma16x16x32(af, bf, acc);
  }
  #pragma unroll
  for (int i = 0; i < 8; ++i) {
    const int co = mt * 16 + i + (hi ? 8 : 0);
    out[((size_t)b * CCH + co) * HWN + t] = acc[i];
  }
}

// ---------------------------------------------------------------------------
extern "C" void kernel_launch(void* const* d_in, const int* in_sizes, int n_in,
                              void* d_out, int out_size, void* d_ws, size_t ws_size,
                              hipStream_t stream) {
  const float* primary_x     = (const float*)d_in[0];
  const float* conditional_x = (const float*)d_in[1];
  const float* convc_w1 = (const float*)d_in[2];
  const float* convc_b1 = (const float*)d_in[3];
  const float* convc_w2 = (const float*)d_in[4];
  const float* convc_b2 = (const float*)d_in[5];
  const float* convp_w1 = (const float*)d_in[6];
  const float* convp_b1 = (const float*)d_in[7];
  const float* convp_w2 = (const float*)d_in[8];
  const float* convp_b2 = (const float*)d_in[9];
  const float* in_proj_w = (const float*)d_in[10];
  const float* conv1d_w  = (const float*)d_in[11];
  const float* conv1d_b  = (const float*)d_in[12];
  const float* x_proj_w  = (const float*)d_in[13];
  const float* dt_proj_w = (const float*)d_in[14];
  const float* dt_proj_b = (const float*)d_in[15];
  const float* A_log     = (const float*)d_in[16];
  const float* D_param   = (const float*)d_in[17];
  const float* out_proj_w = (const float*)d_in[18];
  float* out = (float*)d_out;

  float* ws = (float*)d_ws;
  size_t off = 0;
  float* tmp    = ws + off; off += (size_t)BATCH * CCH * HWN;
  float* stem_c = ws + off; off += (size_t)BATCH * CCH * HWN;
  float* stem_p = ws + off; off += (size_t)BATCH * CCH * HWN;
  float* xz     = ws + off; off += (size_t)BATCH * LTOT * 256;
  float* xc     = ws + off; off += (size_t)BATCH * LTOT * DINNER;
  float* dtb    = ws + off; off += (size_t)BATCH * LTOT * DINNER;
  float* Bm     = ws + off; off += (size_t)BATCH * LTOT * DSTATE;
  float* Cm     = ws + off; off += (size_t)BATCH * LTOT * DSTATE;
  float* aP     = ws + off; off += (size_t)BATCH * NCHUNK * DINNER * DSTATE;
  float* u      = ws + off; off += (size_t)BATCH * NCHUNK * DINNER * DSTATE;
  float* hstart = ws + off; off += (size_t)BATCH * NCHUNK * DINNER * DSTATE;
  float* yf     = ws + off; off += (size_t)BATCH * HWN * DINNER;

  const dim3 blk(256);  // 8 wave32s

  // Conv stems: grid = b(2) x cotg(2) x y(64) = 256 blocks
  conv3x3_leaky_wmma<<<256, blk, 0, stream>>>(conditional_x, convc_w1, convc_b1, tmp);
  conv3x3_leaky_wmma<<<256, blk, 0, stream>>>(tmp, convc_w2, convc_b2, stem_c);
  conv3x3_leaky_wmma<<<256, blk, 0, stream>>>(primary_x, convp_w1, convp_b1, tmp);
  conv3x3_leaky_wmma<<<256, blk, 0, stream>>>(tmp, convp_w2, convp_b2, stem_p);

  // in_proj: grid = b(2) x mtg(2) x nt(512) = 2048 blocks
  in_proj_wmma_k<<<2048, blk, 0, stream>>>(stem_c, stem_p, in_proj_w, xz);

  dwconv_silu<<<(BATCH * LTOT * DINNER) / 256, blk, 0, stream>>>(xz, conv1d_w, conv1d_b, xc);

  xproj_dt<<<BATCH * LTOT, dim3(128), 0, stream>>>(xc, x_proj_w, dt_proj_w, dt_proj_b,
                                                   dtb, Bm, Cm);

  scan_pass1<<<BATCH * NCHUNK * 8, blk, 0, stream>>>(dtb, xc, Bm, A_log, aP, u);
  scan_pass2<<<(BATCH * DINNER * DSTATE) / 256, blk, 0, stream>>>(aP, u, hstart);
  scan_pass3<<<BATCH * 32 * 8, blk, 0, stream>>>(dtb, xc, xz, Bm, Cm, A_log, D_param,
                                                 hstart, yf);

  // out_proj: 2*4*256 = 2048 waves -> 256 blocks, writes NCHW directly
  out_proj_wmma_k<<<256, blk, 0, stream>>>(yf, out_proj_w, out);

  (void)in_sizes; (void)n_in; (void)out_size; (void)ws_size;
}
